// MultiHeadAttention_7825430413765
// MI455X (gfx1250) — matrix operand order repaired and checked by reading, not compiled, not run on hardware
//
#include <hip/hip_runtime.h>
#include <stdint.h>

// MultiHeadAttention forward for MI455X (gfx1250), wave32 + WMMA + TDM.
// B=64, T=256, D_MODEL=512, H=8, D_K=64. All matmuls in f16 WMMA w/ f32 acc.

#define DM   512
#define NH   8
#define DK   64
#define BT   64
#define SEQ  256
#define MROWS (BT*SEQ)   // 16384

typedef __attribute__((ext_vector_type(16))) _Float16 v16h;
typedef __attribute__((ext_vector_type(8)))  _Float16 v8h;
typedef __attribute__((ext_vector_type(4)))  _Float16 v4h;
typedef __attribute__((ext_vector_type(8)))  float    v8f;
typedef __attribute__((ext_vector_type(4)))  float    v4f;
typedef __attribute__((ext_vector_type(4)))  unsigned int u32x4;
typedef __attribute__((ext_vector_type(8)))  int      i32x8;
typedef __attribute__((ext_vector_type(4)))  int      i32x4;

union V16H { v16h v; v8h h8[2]; _Float16 e[16]; };
union V8H  { v8h  v; _Float16 e[8]; };
union V8F  { v8f  v; float e[8]; };

__device__ __forceinline__ v8f wmma16x16x32(v16h a, v16h b, v8f c) {
  // (neg_a, A, neg_b, B, c_mod, C, reuse_a, reuse_b)
  return __builtin_amdgcn_wmma_f32_16x16x32_f16(false, a, false, b, (short)0, c,
                                                false, false);
}

// ---- Tensor Data Mover: 2D tile (tile_d1 rows x tile_d0 elems, f16) -> LDS ----
// Descriptor layout per CDNA5 ISA ch.8 (group0 128b, group1 256b; groups 2/3
// zero for 2D tensors). All inputs are wave-uniform -> SGPR groups.
__device__ __forceinline__ void tdm_load_2d_f16(
    unsigned int lds_addr, const void* gptr,
    unsigned int tensor_d0, unsigned int tensor_d1,
    unsigned int tile_d0, unsigned int tile_d1, unsigned int stride0) {
  unsigned long long ga = (unsigned long long)(uintptr_t)gptr;
  u32x4 g0;
  g0[0] = 1u;                                   // count=1 (valid), user mode
  g0[1] = lds_addr;                             // lds_addr [63:32]
  g0[2] = (unsigned int)ga;                     // global_addr[31:0]
  g0[3] = (unsigned int)((ga >> 32) & 0x1FFFFFFu) | (2u << 30);  // addr hi | type=2
  i32x8 g1;
  g1[0] = 0x00010000;                           // data_size=1 (2B), no flags
  g1[1] = (int)(tensor_d0 << 16);               // barrier_addr=0 | tensor_dim0 lo16
  g1[2] = (int)((tensor_d0 >> 16) | (tensor_d1 << 16));
  g1[3] = (int)((tensor_d1 >> 16) | (tile_d0 << 16));
  g1[4] = (int)(tile_d1 & 0xFFFFu);             // tile_dim1 | tile_dim2=0
  g1[5] = (int)stride0;                         // tensor_dim0_stride lo32
  g1[6] = 0;                                    // stride0 hi | stride1 lo
  g1[7] = 0;
  i32x4 z4 = {0, 0, 0, 0};
#if __has_include(<hip/amd_detail/amd_gfx1250_TDM.h>)
  i32x8 z8 = {};
  __builtin_amdgcn_tensor_load_to_lds(g0, g1, z4, z4, z8, 0);
#else
  __builtin_amdgcn_tensor_load_to_lds(g0, g1, z4, z4, 0);
#endif
}

// ---------------- converters ----------------

__global__ __launch_bounds__(256) void cvt_x_f16(const float* __restrict__ in,
                                                 _Float16* __restrict__ out, int n) {
  int i = (blockIdx.x * 256 + threadIdx.x) * 4;
  if (i < n) {
    v4f x = *(const v4f*)(in + i);
    v4h h; h[0] = (_Float16)x[0]; h[1] = (_Float16)x[1];
    h[2] = (_Float16)x[2]; h[3] = (_Float16)x[3];
    *(v4h*)(out + i) = h;
  }
}

// W[h][c][d] (8x512x64 f32) -> Wt[(h*64+d)][c] (512x512 f16)
__global__ __launch_bounds__(256) void cvt_w_qkv(const float* __restrict__ in,
                                                 _Float16* __restrict__ out) {
  int e = blockIdx.x * 256 + threadIdx.x;   // 262144 elems
  int d = e & 63;
  int c = (e >> 6) & 511;
  int h = e >> 15;
  out[(size_t)(h * DK + d) * DM + c] = (_Float16)in[e];
}

// Wo[k][n] (512x512 f32) -> WoT[n][k] (f16)
__global__ __launch_bounds__(256) void cvt_w_o(const float* __restrict__ in,
                                               _Float16* __restrict__ out) {
  int e = blockIdx.x * 256 + threadIdx.x;   // 262144 elems
  int n = e & 511;
  int k = e >> 9;
  out[(size_t)n * DM + k] = (_Float16)in[e];
}

// ---------------- WMMA GEMM: (16384 x 512) @ (512 x 512) ----------------
// layout 0: f16 out [B,H,T,DK]   (Q, K)
// layout 1: f16 out [B,H,DK,T]   (V transposed)
// layout 2: f32 out row-major MxDM (final projection -> d_out)
__global__ __launch_bounds__(256) void wmma_gemm_512(
    const _Float16* __restrict__ A, const _Float16* __restrict__ Bt,
    _Float16* __restrict__ outH, float* __restrict__ outF, int layout) {
  const int lane = threadIdx.x & 31;
  const int wave = threadIdx.x >> 5;
  const int l15  = lane & 15;
  const int hid  = lane >> 4;
  const int mBase = blockIdx.x * 128 + wave * 16;  // 8 waves x 16 rows
  const int nBase = blockIdx.y * 64;               // 4 n-tiles of 16

  const _Float16* aRow = A + (size_t)(mBase + l15) * DM;
  const _Float16* bRow = Bt + (size_t)(nBase + l15) * DM;

  V8F acc[4] = {};
  for (int k0 = 0; k0 < DM; k0 += 32) {
    __builtin_prefetch((const void*)(aRow + k0 + 128), 0, 1);
    V16H ua;
    ua.h8[0] = *(const v8h*)(aRow + k0 + 16 * hid);       // k: 0..7  / 16..23
    ua.h8[1] = *(const v8h*)(aRow + k0 + 16 * hid + 8);   // k: 8..15 / 24..31
    v16h a = ua.v;
#pragma unroll
    for (int j = 0; j < 4; ++j) {
      v16h b = *(const v16h*)(bRow + (size_t)j * 16 * DM + k0 + 16 * hid);
      acc[j].v = wmma16x16x32(a, b, acc[j].v);
    }
  }

#pragma unroll
  for (int j = 0; j < 4; ++j) {
#pragma unroll
    for (int r = 0; r < 8; ++r) {
      int row = mBase + r + 8 * hid;        // C layout: lanes16-31 -> M = r+8
      int col = nBase + j * 16 + l15;
      float v = acc[j].e[r];
      if (layout == 2) {
        outF[(size_t)row * DM + col] = v;
      } else {
        int bb = row >> 8, t = row & 255;
        int hh = col >> 6, d = col & 63;
        size_t idx = (layout == 0)
            ? (((size_t)(bb * NH + hh) * SEQ + t) * DK + d)
            : (((size_t)(bb * NH + hh) * DK + d) * SEQ + t);
        outH[idx] = (_Float16)v;
      }
    }
  }
}

// ---------------- causal flash attention (TDM-staged K/V) ----------------
// One wave per (b, h, 16-row q-tile). Q/K row-major [bh][t][d], V transposed
// [bh][d][t]. K/V blocks DMA'd to LDS by the Tensor Data Mover.
__global__ __launch_bounds__(32) void attn_fwd(
    const _Float16* __restrict__ Q, const _Float16* __restrict__ K,
    const _Float16* __restrict__ Vt, _Float16* __restrict__ O) {
  __shared__ _Float16 kTile[32 * 64];  // K rows s0..s0+31, row-major [s][d]
  __shared__ _Float16 vTile[64 * 32];  // Vt rows d=0..63, cols s0..s0+31 [d][s]
  __shared__ _Float16 pTile[32 * 16];  // P tile, col-major: [s-col][q-row]

  const int lane = threadIdx.x;
  const int l15  = lane & 15;
  const int hid  = lane >> 4;
  const int bid  = blockIdx.x;
  const int qt   = bid & 15;          // T/16 tiles
  const int h    = (bid >> 4) & 7;
  const int b    = bid >> 7;
  const int tq0  = qt << 4;

  const _Float16* Qbh = Q  + (size_t)(b * NH + h) * SEQ * DK;
  const _Float16* Kbh = K  + (size_t)(b * NH + h) * SEQ * DK;
  const _Float16* Vbh = Vt + (size_t)(b * NH + h) * DK * SEQ;

  const unsigned int kLds = (unsigned int)(uintptr_t)(void*)kTile;
  const unsigned int vLds = (unsigned int)(uintptr_t)(void*)vTile;

  // Q tile as two A-operands (k-dim d: 0..31 and 32..63)
  const _Float16* qRow = Qbh + (size_t)(tq0 + l15) * DK;
  V16H u0, u1;
  u0.h8[0] = *(const v8h*)(qRow + 16 * hid);
  u0.h8[1] = *(const v8h*)(qRow + 16 * hid + 8);
  u1.h8[0] = *(const v8h*)(qRow + 32 + 16 * hid);
  u1.h8[1] = *(const v8h*)(qRow + 32 + 16 * hid + 8);
  const v16h qa0 = u0.v, qa1 = u1.v;

  const float scale = 0.04419417382415922f;  // 1/sqrt(D_MODEL)
  float mrow[8], lrow[8];
  V8F o[4] = {};
#pragma unroll
  for (int r = 0; r < 8; ++r) { mrow[r] = -1e30f; lrow[r] = 0.f; }

  for (int s0 = 0; s0 < tq0 + 16; s0 += 32) {
    // WAR guard: previous iter's ds reads must drain before TDM rewrites LDS.
    __asm__ volatile("s_wait_dscnt 0x0" ::: "memory");
    // K block: 32 rows x 64 cols from K[bh] (row stride 64)
    tdm_load_2d_f16(kLds, (const void*)(Kbh + (size_t)s0 * DK),
                    DK, SEQ, DK, 32, DK);
    // V block: 64 rows x 32 cols from Vt[bh] (row stride 256)
    tdm_load_2d_f16(vLds, (const void*)(Vbh + s0),
                    SEQ, DK, 32, DK, SEQ);
    __builtin_amdgcn_s_wait_tensorcnt(0);
    __asm__ volatile("" ::: "memory");

    // ---- scores: two 16x16 tiles (cols s0..s0+15, s0+16..s0+31) ----
    V8F sc0 = {}, sc1 = {};
    {
      const _Float16* kR0 = &kTile[(size_t)l15 * DK];
      const _Float16* kR1 = &kTile[(size_t)(16 + l15) * DK];
      sc0.v = wmma16x16x32(qa0, *(const v16h*)(kR0 + 16 * hid), sc0.v);
      sc0.v = wmma16x16x32(qa1, *(const v16h*)(kR0 + 32 + 16 * hid), sc0.v);
      sc1.v = wmma16x16x32(qa0, *(const v16h*)(kR1 + 16 * hid), sc1.v);
      sc1.v = wmma16x16x32(qa1, *(const v16h*)(kR1 + 32 + 16 * hid), sc1.v);
    }

    // ---- scale + causal mask + running row max ----
    float rmax[8];
#pragma unroll
    for (int r = 0; r < 8; ++r) {
      int trow = tq0 + r + 8 * hid;
      float v0 = sc0.e[r] * scale;
      float v1 = sc1.e[r] * scale;
      if (s0 + l15 > trow)      v0 = -1e30f;
      if (s0 + 16 + l15 > trow) v1 = -1e30f;
      sc0.e[r] = v0; sc1.e[r] = v1;
      rmax[r] = fmaxf(v0, v1);
    }
#pragma unroll
    for (int off = 1; off < 16; off <<= 1)
#pragma unroll
      for (int r = 0; r < 8; ++r)
        rmax[r] = fmaxf(rmax[r], __shfl_xor(rmax[r], off, 32));

    // ---- online softmax update ----
    float corr[8], rsum[8];
#pragma unroll
    for (int r = 0; r < 8; ++r) {
      float nm = fmaxf(mrow[r], rmax[r]);
      corr[r] = __expf(mrow[r] - nm);
      mrow[r] = nm;
      float p0 = __expf(sc0.e[r] - nm);
      float p1 = __expf(sc1.e[r] - nm);
      sc0.e[r] = p0; sc1.e[r] = p1;
      rsum[r] = p0 + p1;
    }
#pragma unroll
    for (int off = 1; off < 16; off <<= 1)
#pragma unroll
      for (int r = 0; r < 8; ++r)
        rsum[r] += __shfl_xor(rsum[r], off, 32);
#pragma unroll
    for (int r = 0; r < 8; ++r) lrow[r] = lrow[r] * corr[r] + rsum[r];
#pragma unroll
    for (int j = 0; j < 4; ++j)
#pragma unroll
      for (int r = 0; r < 8; ++r) o[j].e[r] *= corr[r];

    // ---- P (C-layout f32) -> LDS -> A-operand f16 ----
    V8H p0h, p1h;
#pragma unroll
    for (int r = 0; r < 8; ++r) {
      p0h.e[r] = (_Float16)sc0.e[r];
      p1h.e[r] = (_Float16)sc1.e[r];
    }
    // col-major LDS: column index = s-col, rows r+8*hid are contiguous
    *(v8h*)&pTile[(l15) * 16 + 8 * hid]      = p0h.v;
    *(v8h*)&pTile[(16 + l15) * 16 + 8 * hid] = p1h.v;

    V16H pa;
#pragma unroll
    for (int i = 0; i < 8; ++i) {
      pa.e[i]     = pTile[(16 * hid + i) * 16 + l15];       // k 16*hid + 0..7
      pa.e[8 + i] = pTile[(16 * hid + 8 + i) * 16 + l15];   // k 16*hid + 8..15
    }

    // ---- P @ V : 4 d-tiles (B-operand from LDS-staged V) ----
#pragma unroll
    for (int j = 0; j < 4; ++j) {
      const _Float16* vp = &vTile[(size_t)(j * 16 + l15) * 32 + 16 * hid];
      o[j].v = wmma16x16x32(pa.v, *(const v16h*)vp, o[j].v);
    }
  }

  // ---- normalize + store [b][t][h*64+d] ----
#pragma unroll
  for (int j = 0; j < 4; ++j)
#pragma unroll
    for (int r = 0; r < 8; ++r) {
      int trow = tq0 + r + 8 * hid;
      int d = j * 16 + l15;
      float val = o[j].e[r] / lrow[r];
      O[((size_t)(b * SEQ + trow)) * DM + h * DK + d] = (_Float16)val;
    }
}

// ---------------- launch ----------------

extern "C" void kernel_launch(void* const* d_in, const int* in_sizes, int n_in,
                              void* d_out, int out_size, void* d_ws, size_t ws_size,
                              hipStream_t stream) {
  const float* x  = (const float*)d_in[0];
  const float* Wq = (const float*)d_in[1];
  const float* Wk = (const float*)d_in[2];
  const float* Wv = (const float*)d_in[3];
  const float* Wo = (const float*)d_in[4];
  float* out = (float*)d_out;

  const size_t XE = (size_t)MROWS * DM;   // 8388608 elems
  const size_t WE = (size_t)DM * DM;      // 262144 elems
  _Float16* ws  = (_Float16*)d_ws;
  _Float16* xh  = ws;            // x f16            (16 MB)
  _Float16* WqT = xh  + XE;      // Wq^T f16
  _Float16* WkT = WqT + WE;
  _Float16* WvT = WkT + WE;
  _Float16* WoT = WvT + WE;
  _Float16* Qh  = WoT + WE;      // [B,H,T,DK] f16
  _Float16* Kh  = Qh  + XE;      // [B,H,T,DK] f16
  _Float16* Vth = Kh  + XE;      // [B,H,DK,T] f16
  _Float16* Ah  = Vth + XE;      // attn out [B,T,H*DK] f16

  cvt_x_f16<<<(int)(XE / (256 * 4)), 256, 0, stream>>>(x, xh, (int)XE);
  cvt_w_qkv<<<(int)(WE / 256), 256, 0, stream>>>(Wq, WqT);
  cvt_w_qkv<<<(int)(WE / 256), 256, 0, stream>>>(Wk, WkT);
  cvt_w_qkv<<<(int)(WE / 256), 256, 0, stream>>>(Wv, WvT);
  cvt_w_o  <<<(int)(WE / 256), 256, 0, stream>>>(Wo, WoT);

  dim3 gGemm(MROWS / 128, DM / 64);
  wmma_gemm_512<<<gGemm, 256, 0, stream>>>(xh, WqT, Qh,  nullptr, 0);
  wmma_gemm_512<<<gGemm, 256, 0, stream>>>(xh, WkT, Kh,  nullptr, 0);
  wmma_gemm_512<<<gGemm, 256, 0, stream>>>(xh, WvT, Vth, nullptr, 1);

  attn_fwd<<<BT * NH * (SEQ / 16), 32, 0, stream>>>(Qh, Kh, Vth, Ah);

  wmma_gemm_512<<<gGemm, 256, 0, stream>>>(Ah, WoT, nullptr, out, 2);
}
